// CausalSelfAttention_38053410242838
// MI455X (gfx1250) — matrix operand order repaired and checked by reading, not compiled, not run on hardware
//
#include <hip/hip_runtime.h>

#define DEV __device__ __forceinline__

typedef __attribute__((ext_vector_type(8)))  float  v8f;
typedef __attribute__((ext_vector_type(8)))  __bf16 v8bf;
typedef __attribute__((ext_vector_type(16))) __bf16 v16bf;

constexpr int BB = 2, T = 2048, C = 1024, NH = 16, HD = 64;
constexpr int M  = BB * T;      // 4096 rows (B*T)
constexpr int KD = C;           // 1024 reduction dim
constexpr int N1 = 3 * C;       // 3072 qkv columns

struct BF16Frag { v8bf lo, hi; };

static DEV v16bf frag_join(v8bf lo, v8bf hi) {
  BF16Frag t{lo, hi};
  return __builtin_bit_cast(v16bf, t);
}

// fp32 -> bf16; native conversion (RNE) so the backend can use v_cvt[_pk]
static DEV __bf16 f2bf(float f) { return (__bf16)f; }

static DEV v8f vzero() {
  v8f z = {0.f, 0.f, 0.f, 0.f, 0.f, 0.f, 0.f, 0.f};
  return z;
}

// A fragment, 16xK row-major source. ISA 16-bit A layout:
// lanes 0-15: row = lane, K chunks {k0..k0+7} and {k0+16..k0+23}
// lanes 16-31: row = lane-16, K chunks {k0+8..k0+15} and {k0+24..k0+31}
static DEV v16bf load_a_rm(const __bf16* A, int lda, int lane, int k0) {
  int m  = lane & 15;
  int kl = k0 + ((lane & 16) ? 8 : 0);
  const __bf16* p = A + m * lda + kl;
  v8bf lo = *reinterpret_cast<const v8bf*>(p);
  v8bf hi = *reinterpret_cast<const v8bf*>(p + 16);
  return frag_join(lo, hi);
}

// B fragment for C = A * W^T with W row-major (N x K).
// ISA B layout: lane holds column N = lane%16; lanes 0-15 hold K chunks {k0..k0+7} and {k0+16..k0+23},
// lanes 16-31 hold K chunks {k0+8..k0+15} and {k0+24..k0+31}, consecutive K packed 2/VGPR.
static DEV v16bf load_b_tn(const __bf16* Wm, int ldb, int lane, int k0) {
  int n  = lane & 15;
  int kb = k0 + ((lane & 16) ? 8 : 0);
  const __bf16* p = Wm + n * ldb + kb;
  v8bf lo = *reinterpret_cast<const v8bf*>(p);
  v8bf hi = *reinterpret_cast<const v8bf*>(p + 16);
  return frag_join(lo, hi);
}

static DEV v8f wmma_bf16(v16bf a, v16bf b, v8f c) {
  return __builtin_amdgcn_wmma_f32_16x16x32_bf16(
      /*neg_a=*/false, a, /*neg_b=*/false, b,
      /*c_mod=*/(short)0, c, /*reuse_a=*/false, /*reuse_b=*/false);
}

// ---------------------------------------------------------------- cvt ------
__global__ __launch_bounds__(256) void cvt_f32_bf16(const float* __restrict__ src,
                                                    __bf16* __restrict__ dst, int n) {
  for (int i = blockIdx.x * blockDim.x + threadIdx.x; i < n;
       i += gridDim.x * blockDim.x)
    dst[i] = f2bf(src[i]);
}

// Shared 32(M) x 64(N) x KD mainloop: acc[mi][nt], B fragments reused 2x.
static DEV void gemm_main_32x64(const __bf16* __restrict__ A, int m0,
                                const __bf16* __restrict__ W, int n0,
                                int lane, v8f acc[2][4]) {
#pragma unroll
  for (int mi = 0; mi < 2; ++mi)
#pragma unroll
    for (int t = 0; t < 4; ++t) acc[mi][t] = vzero();

  const __bf16* A0 = A + (size_t)m0 * KD;
  const __bf16* A1 = A0 + (size_t)16 * KD;
  for (int k0 = 0; k0 < KD; k0 += 32) {
    v16bf af0 = load_a_rm(A0, KD, lane, k0);
    v16bf af1 = load_a_rm(A1, KD, lane, k0);
#pragma unroll
    for (int t = 0; t < 4; ++t) {
      v16bf bf = load_b_tn(W + (size_t)(n0 + t * 16) * KD, KD, lane, k0);
      acc[0][t] = wmma_bf16(af0, bf, acc[0][t]);
      acc[1][t] = wmma_bf16(af1, bf, acc[1][t]);
    }
  }
}

// ------------------------------------------------------------ qkv GEMM ----
// kqv = x @ W_kqv^T + b ; scatter into K (b,h,t,d), Q (b,h,t,d), V^T (b,h,d,t)
__global__ __launch_bounds__(256) void gemm_kqv(const __bf16* __restrict__ A,
                                                const __bf16* __restrict__ W,
                                                const float* __restrict__ bias,
                                                __bf16* __restrict__ Qb,
                                                __bf16* __restrict__ Kb,
                                                __bf16* __restrict__ VTb) {
  int lane = threadIdx.x & 31;
  int w = blockIdx.x * 8 + (threadIdx.x >> 5);
  constexpr int NT = N1 / 64;  // 48
  int mt = w / NT, nt = w % NT;
  int m0 = mt * 32, n0 = nt * 64;

  v8f acc[2][4];
  gemm_main_32x64(A, m0, W, n0, lane, acc);

  int hi = lane >> 4, col = lane & 15;
#pragma unroll
  for (int t = 0; t < 4; ++t) {
    int n = n0 + t * 16 + col;
    float bv = bias[n];
    int h = n / (3 * HD);
    int r = n % (3 * HD);
    int s = r / HD;
    int d = r % HD;
#pragma unroll
    for (int mi = 0; mi < 2; ++mi)
#pragma unroll
      for (int j = 0; j < 8; ++j) {
        int m = m0 + mi * 16 + j + 8 * hi;
        int b = m >> 11;          // m / T
        int tt = m & (T - 1);     // m % T
        __bf16 v = f2bf(acc[mi][t][j] + bv);
        size_t bh = (size_t)b * NH + h;
        if (s == 0)      Kb [(bh * T + tt) * HD + d] = v;
        else if (s == 1) Qb [(bh * T + tt) * HD + d] = v;
        else             VTb[(bh * HD + d) * T + tt] = v;
      }
  }
}

// ---------------------------------------------------------- attention -----
// One wave per 16-row q tile of one (b,h); 64 k-columns per iteration.
__global__ __launch_bounds__(256) void attn_flash(const __bf16* __restrict__ Qb,
                                                  const __bf16* __restrict__ Kb,
                                                  const __bf16* __restrict__ VTb,
                                                  __bf16* __restrict__ Yb) {
  __shared__ alignas(16) __bf16 Pb[8][16 * 64];
  int lane = threadIdx.x & 31;
  int wv = threadIdx.x >> 5;
  int w = blockIdx.x * 8 + wv;
  int qt = w & 127;             // T/16 == 128 tiles
  int bh = w >> 7;
  int tq0 = qt * 16;
  int b = bh >> 4, h = bh & 15;

  const __bf16* Qbase = Qb + (size_t)bh * T * HD + (size_t)tq0 * HD;
  const __bf16* Kbase = Kb + (size_t)bh * T * HD;
  const __bf16* Vbase = VTb + (size_t)bh * HD * T;
  __bf16* myP = &Pb[wv][0];

  int hi = lane >> 4, col = lane & 15;

  float m_run[8], l_run[8];
#pragma unroll
  for (int j = 0; j < 8; ++j) { m_run[j] = -3.0e38f; l_run[j] = 0.f; }
  v8f o[4];
#pragma unroll
  for (int t = 0; t < 4; ++t) o[t] = vzero();

  v16bf q0 = load_a_rm(Qbase, HD, lane, 0);
  v16bf q1 = load_a_rm(Qbase, HD, lane, 32);

  const float scale = 0.125f;   // 1/sqrt(64)
  int smax = tq0 + 15;
  for (int s0 = 0; s0 <= smax; s0 += 64) {
    __builtin_prefetch(Kbase + (size_t)(s0 + 64) * HD, 0, 1);

    // S = q k^T over 64 columns (4 n-subtiles, K-dim = 64 split in 2)
    v8f s[4];
#pragma unroll
    for (int ti = 0; ti < 4; ++ti) {
      s[ti] = vzero();
      const __bf16* kp = Kbase + (size_t)(s0 + ti * 16) * HD;
      s[ti] = wmma_bf16(q0, load_b_tn(kp, HD, lane, 0), s[ti]);
      s[ti] = wmma_bf16(q1, load_b_tn(kp, HD, lane, 32), s[ti]);
    }

    // scale + causal mask (C layout: row = j + 8*hi, col = lane%16)
    float mt[8];
#pragma unroll
    for (int j = 0; j < 8; ++j) {
      int tq = tq0 + j + 8 * hi;
      mt[j] = -3.0e38f;
#pragma unroll
      for (int ti = 0; ti < 4; ++ti) {
        int tk = s0 + ti * 16 + col;
        float v = s[ti][j] * scale;
        s[ti][j] = (tk <= tq) ? v : -3.0e38f;
        mt[j] = fmaxf(mt[j], s[ti][j]);
      }
    }
    // row max across the 16 lanes holding the row (stays within half-wave)
#pragma unroll
    for (int off = 8; off >= 1; off >>= 1)
#pragma unroll
      for (int j = 0; j < 8; ++j)
        mt[j] = fmaxf(mt[j], __shfl_xor(mt[j], off, 32));

    float rs[8], cf[8];
#pragma unroll
    for (int j = 0; j < 8; ++j) {
      float mn = fmaxf(m_run[j], mt[j]);
      cf[j] = __expf(m_run[j] - mn);
      m_run[j] = mn;
      rs[j] = 0.f;
#pragma unroll
      for (int ti = 0; ti < 4; ++ti) {
        float p = __expf(s[ti][j] - mn);
        s[ti][j] = p;
        rs[j] += p;
      }
    }
#pragma unroll
    for (int off = 8; off >= 1; off >>= 1)
#pragma unroll
      for (int j = 0; j < 8; ++j)
        rs[j] += __shfl_xor(rs[j], off, 32);
#pragma unroll
    for (int j = 0; j < 8; ++j)
      l_run[j] = l_run[j] * cf[j] + rs[j];
#pragma unroll
    for (int t = 0; t < 4; ++t)
#pragma unroll
      for (int j = 0; j < 8; ++j)
        o[t][j] *= cf[j];

    // restage P (16x64) row-major in LDS, then reload as WMMA A fragments
#pragma unroll
    for (int ti = 0; ti < 4; ++ti)
#pragma unroll
      for (int j = 0; j < 8; ++j)
        myP[(j + 8 * hi) * 64 + ti * 16 + col] = f2bf(s[ti][j]);
    asm volatile("s_wait_dscnt 0" ::: "memory");  // intra-wave cross-lane LDS dep

    v16bf pf0 = load_a_rm(myP, 64, lane, 0);
    v16bf pf1 = load_a_rm(myP, 64, lane, 32);
    __builtin_prefetch(Vbase + s0 + 64, 0, 1);
#pragma unroll
    for (int dt = 0; dt < 4; ++dt) {
      const __bf16* vp = Vbase + (size_t)(dt * 16) * T + s0;  // V^T: contiguous K
      o[dt] = wmma_bf16(pf0, load_b_tn(vp, T, lane, 0), o[dt]);
      o[dt] = wmma_bf16(pf1, load_b_tn(vp, T, lane, 32), o[dt]);
    }
  }

  float inv[8];
#pragma unroll
  for (int j = 0; j < 8; ++j) inv[j] = 1.0f / l_run[j];
#pragma unroll
  for (int dt = 0; dt < 4; ++dt)
#pragma unroll
    for (int j = 0; j < 8; ++j) {
      int row = b * T + tq0 + j + 8 * hi;
      int c = h * HD + dt * 16 + col;
      Yb[(size_t)row * C + c] = f2bf(o[dt][j] * inv[j]);
    }
}

// ----------------------------------------------------------- out GEMM -----
__global__ __launch_bounds__(256) void gemm_proj(const __bf16* __restrict__ A,
                                                 const __bf16* __restrict__ W,
                                                 const float* __restrict__ bias,
                                                 float* __restrict__ out) {
  int lane = threadIdx.x & 31;
  int w = blockIdx.x * 8 + (threadIdx.x >> 5);
  constexpr int NT = C / 64;  // 16
  int mt = w / NT, nt = w % NT;
  int m0 = mt * 32, n0 = nt * 64;

  v8f acc[2][4];
  gemm_main_32x64(A, m0, W, n0, lane, acc);

  int hi = lane >> 4, col = lane & 15;
#pragma unroll
  for (int t = 0; t < 4; ++t) {
    int n = n0 + t * 16 + col;
    float bv = bias[n];
#pragma unroll
    for (int mi = 0; mi < 2; ++mi)
#pragma unroll
      for (int j = 0; j < 8; ++j) {
        int m = m0 + mi * 16 + j + 8 * hi;
        out[(size_t)m * C + n] = acc[mi][t][j] + bv;
      }
  }
}

// --------------------------------------------------------------------------
extern "C" void kernel_launch(void* const* d_in, const int* in_sizes, int n_in,
                              void* d_out, int out_size, void* d_ws, size_t ws_size,
                              hipStream_t stream) {
  const float* x     = (const float*)d_in[0];
  const float* Wkqv  = (const float*)d_in[1];
  const float* bkqv  = (const float*)d_in[2];
  const float* Wproj = (const float*)d_in[3];
  const float* bproj = (const float*)d_in[4];
  float* out = (float*)d_out;

  char* ws = (char*)d_ws;
  const size_t MB = 1024 * 1024;
  __bf16* xb     = (__bf16*)(ws + 0 * MB);   // 8 MB  (4096x1024)
  __bf16* wkqvb  = (__bf16*)(ws + 8 * MB);   // 6 MB  (3072x1024)
  __bf16* wprojb = (__bf16*)(ws + 14 * MB);  // 2 MB  (1024x1024)
  __bf16* Qb     = (__bf16*)(ws + 16 * MB);  // 8 MB  (b,h,t,d)
  __bf16* Kb     = (__bf16*)(ws + 24 * MB);  // 8 MB  (b,h,t,d)
  __bf16* VTb    = (__bf16*)(ws + 32 * MB);  // 8 MB  (b,h,d,t)
  __bf16* Yb     = (__bf16*)(ws + 40 * MB);  // 8 MB  (4096x1024)

  cvt_f32_bf16<<<2048, 256, 0, stream>>>(x, xb, M * C);
  cvt_f32_bf16<<<2048, 256, 0, stream>>>(Wkqv, wkqvb, N1 * C);
  cvt_f32_bf16<<<1024, 256, 0, stream>>>(Wproj, wprojb, C * C);

  gemm_kqv<<<(M / 32) * (N1 / 64) / 8, 256, 0, stream>>>(xb, wkqvb, bkqv,
                                                         Qb, Kb, VTb);
  attn_flash<<<(BB * NH * (T / 16)) / 8, 256, 0, stream>>>(Qb, Kb, VTb, Yb);
  gemm_proj<<<(M / 32) * (C / 64) / 8, 256, 0, stream>>>(Yb, wprojb, bproj, out);
}
